// PSGNet_63900523430216
// MI455X (gfx1250) — matrix operand order repaired and checked by reading, not compiled, not run on hardware
//
#include <hip/hip_runtime.h>
#include <hip/hip_bf16.h>
#include <stdint.h>

typedef unsigned short u16;
typedef __attribute__((ext_vector_type(16))) __bf16 v16bf;
typedef __attribute__((ext_vector_type(8)))  float  v8f;

#define NPIX 65536   // 16 * 64 * 64

// ---------- bf16 helpers (RNE) ----------
__device__ __forceinline__ u16 f2bf(float f) {
  unsigned u = __float_as_uint(f);
  u += 0x7fffu + ((u >> 16) & 1u);
  return (u16)(u >> 16);
}
__device__ __forceinline__ float bf2f(u16 h) {
  return __uint_as_float(((unsigned)h) << 16);
}
// order-preserving float<->uint for atomicMax on floats
__device__ __forceinline__ unsigned encf(float f) {
  unsigned u = __float_as_uint(f);
  return (u & 0x80000000u) ? ~u : (u | 0x80000000u);
}
__device__ __forceinline__ float decf(unsigned u) {
  return __uint_as_float((u & 0x80000000u) ? (u & 0x7fffffffu) : ~u);
}

// ---------- NHWC fp32 image -> [3][NPIX] bf16 ----------
__global__ void img2act_kernel(const float* __restrict__ img, u16* __restrict__ out) {
  int idx = blockIdx.x * blockDim.x + threadIdx.x;
  if (idx >= 3 * NPIX) return;
  int c = idx / NPIX;
  int p = idx - c * NPIX;          // p = ((b*64+y)*64+x); img flat = p*3 + c
  out[idx] = f2bf(img[p * 3 + c]);
}

// ---------- pack conv weights [Cout][Cin][ks][ks] into WMMA A-matrix lane layout ----------
// block layout: [mt][kt][lane(32)][elem(16)]  (bf16), K = Cin*ks*ks padded to 32
__global__ void pack_w_kernel(const float* __restrict__ w, u16* __restrict__ out,
                              int Cout, int Cin, int ks, int total) {
  int idx = blockIdx.x * blockDim.x + threadIdx.x;
  if (idx >= total) return;
  int K  = Cin * ks * ks;
  int KT = (K + 31) >> 5;
  int e  = idx & 15;
  int l  = (idx >> 4) & 31;
  int bt = idx >> 9;               // mt*KT + kt
  int kt = bt % KT;
  int mt = bt / KT;
  int m     = mt * 16 + (l & 15);
  int khalf = l >> 4;
  // 16-bit A-matrix 16x32 layout (ISA 7.12.2): lane<16 -> K {0..7,16..23}, lane>=16 -> K {8..15,24..31}
  int k = kt * 32 + khalf * 8 + ((e < 8) ? e : e + 8);
  float v = 0.f;
  if (m < Cout && k < K) v = w[m * K + k];
  out[idx] = f2bf(v);
}

// ---------- conv as implicit GEMM via v_wmma_f32_16x16x32_bf16 ----------
// One 128-thread block handles one full image row (64 px) for one 16-channel
// output tile. Input rows (with y-halo) are staged into LDS with the CDNA5
// async global->LDS copy engine (ASYNCcnt-tracked, no VGPR round-trip), then
// each of the 4 waves computes a 16(ch)x16(px) WMMA tile, K fully unrolled.
// in   : bf16 [CIN][NPIX]  (channel-major, p = b*4096 + y*64 + x)
// wpack: A-matrix blocks as packed above
// out  : bf16 [Cout][NPIX] (pointer may be pre-offset for concat writes)
template<int KS, int CIN>
__global__ __launch_bounds__(128) void conv_wmma_t(
    const u16* __restrict__ in, const u16* __restrict__ wpack,
    const float* __restrict__ bias, u16* __restrict__ out,
    const u16* __restrict__ res, float* __restrict__ nodes,
    int Cout, int relu) {
  constexpr int K    = CIN * KS * KS;
  constexpr int KT   = (K + 31) / 32;
  constexpr int ROWS = CIN * KS;            // KS stacked rows per input channel
  __shared__ u16 smem[ROWS * 64];

  const int lane = threadIdx.x & 31;
  const int wave = threadIdx.x >> 5;
  const int b  = blockIdx.x >> 6;           // image
  const int y  = blockIdx.x & 63;           // row
  const int mt = blockIdx.y;

  // ---- stage CIN*KS rows of 64 bf16 (128B each) into LDS via async copy ----
  {
    const unsigned* in32 = (const unsigned*)in;
    unsigned* s32 = (unsigned*)smem;
    const int baseoff = b << 12;
    for (int w = threadIdx.x; w < ROWS * 32; w += 128) {
      int row = w >> 5;                     // cin*KS + ky
      int wd  = w & 31;
      int cin = (KS == 3) ? (row / 3) : row;
      int ky  = (KS == 3) ? (row - cin * 3) : 0;
      int yy  = y + ky - (KS >> 1);
      unsigned ldsaddr = (unsigned)(uintptr_t)(&s32[w]);   // LDS byte offset
      if ((unsigned)yy < 64u) {
        const unsigned* gptr =
            in32 + (((size_t)cin * NPIX + baseoff + (yy << 6)) >> 1) + wd;
        // GLOBAL_LOAD_ASYNC_TO_LDS_B32: LDS[vdst] = MEM[vaddr] (ASYNCcnt)
        asm volatile("global_load_async_to_lds_b32 %0, %1, off"
                     :: "v"(ldsaddr), "v"(gptr) : "memory");
      } else {
        s32[w] = 0u;                        // zero-fill y-halo
      }
    }
  }
  asm volatile("s_wait_asynccnt 0x0" ::: "memory");
  __syncthreads();

  const int n     = lane & 15;
  const int khalf = lane >> 4;
  const int x     = (wave << 4) + n;
  const int p     = (blockIdx.x << 6) + x;

  v8f acc = {};
  const u16* wp = wpack + (size_t)mt * KT * 512 + lane * 16;
#pragma unroll
  for (int kt = 0; kt < KT; ++kt) {
    // A fragment: contiguous 16 bf16 per lane, pre-swizzled -> 2x b128 loads
    v16bf afrag = *(const v16bf*)(wp + (size_t)kt * 512);
    if (kt + 1 < KT) __builtin_prefetch(wp + (size_t)(kt + 1) * 512, 0, 1);
    // B fragment: 16-bit B 32x16 layout: lane half -> K {0..7,16..23} / {8..15,24..31}, N = lane&15
    v16bf bfrag;
#pragma unroll
    for (int e = 0; e < 16; ++e) {
      int k = kt * 32 + khalf * 8 + ((e < 8) ? e : e + 8);     // divides below are by constants
      u16 v = 0;
      if (KS == 3) {
        int row = k / 3;                    // = cin*3 + ky (k = cin*9+ky*3+kx)
        int kx  = k - row * 3;
        int xx  = x + kx - 1;
        if (k < K && (unsigned)xx < 64u) v = smem[row * 64 + xx];
      } else {
        if (k < K) v = smem[k * 64 + x];
      }
      bfrag[e] = __builtin_bit_cast(__bf16, v);
    }
    acc = __builtin_amdgcn_wmma_f32_16x16x32_bf16(
        /*neg_a=*/false, afrag, /*neg_b=*/false, bfrag,
        /*c_mod=*/(short)0, acc, /*reuse_a=*/false, /*reuse_b=*/false);
  }
  // epilogue: C/D layout (ISA 7.12.2): VGPR r -> M = r + 8*(lane>=16), N = lane&15
#pragma unroll
  for (int r = 0; r < 8; ++r) {
    int m = mt * 16 + r + (khalf << 3);
    if (m < Cout) {
      float v = acc[r] + bias[m];
      if (relu) v = fmaxf(v, 0.f);
      if (res) v += bf2f(res[(size_t)m * NPIX + p]);
      out[(size_t)m * NPIX + p] = f2bf(v);
      if (nodes) nodes[(size_t)p * 32 + m] = v;
    }
  }
}

// ---------- graph stage kernels ----------
__global__ void aff_kernel(const float* __restrict__ nodes, const int* __restrict__ src,
                           const int* __restrict__ dst, float* __restrict__ aff,
                           float* __restrict__ sumaff, int* __restrict__ degi, int E) {
  int e = blockIdx.x * blockDim.x + threadIdx.x;
  if (e >= E) return;
  int s = src[e], d = dst[e];
  const float* a = nodes + (size_t)s * 32;
  const float* c = nodes + (size_t)d * 32;
  float acc = 0.f;
#pragma unroll
  for (int i = 0; i < 32; ++i) { float t = a[i] - c[i]; acc += t * t; }
  float v = sqrtf(acc);
  aff[e] = v;
  atomicAdd(&sumaff[s], v);
  atomicAdd(&degi[s], 1);
}

__global__ void mean_kernel(const float* __restrict__ sumaff, const int* __restrict__ degi,
                            float* __restrict__ meanv) {
  int nidx = blockIdx.x * blockDim.x + threadIdx.x;
  if (nidx >= NPIX) return;
  meanv[nidx] = sumaff[nidx] / fmaxf((float)degi[nidx], 1.f);
}

__global__ void keep_kernel(const float* __restrict__ aff, const float* __restrict__ meanv,
                            const int* __restrict__ src, const int* __restrict__ dst,
                            unsigned* __restrict__ nbrmask, int E) {
  int e = blockIdx.x * blockDim.x + threadIdx.x;
  if (e >= E) return;
  int s = src[e], d = dst[e];
  if (aff[e] <= fminf(meanv[s], meanv[d])) {
    int ys = (s >> 6) & 63, xs = s & 63;
    int yd = (d >> 6) & 63, xd = d & 63;
    int dirid = (ys - yd + 1) * 3 + (xs - xd + 1);  // 0..8, !=4 by construction
    atomicOr(&nbrmask[d], 1u << dirid);
  }
}

__global__ void iota_kernel(int* __restrict__ lab) {
  int nidx = blockIdx.x * blockDim.x + threadIdx.x;
  if (nidx < NPIX) lab[nidx] = nidx;
}

// one LP step: mode of kept in-neighbor labels; tie -> smallest; none kept -> keep own
__global__ void lp_kernel(const int* __restrict__ lin, int* __restrict__ lout,
                          const unsigned* __restrict__ nmask) {
  int nidx = blockIdx.x * blockDim.x + threadIdx.x;
  if (nidx >= NPIX) return;
  unsigned m = nmask[nidx];
  int b = nidx >> 12, y = (nidx >> 6) & 63, x = nidx & 63;
  int labs[8];
  int cnt = 0;
#pragma unroll
  for (int d = 0; d < 9; ++d) {
    if (d == 4) continue;
    if ((m >> d) & 1u) {
      int oy = d / 3 - 1, ox = d % 3 - 1;
      int sn = (b << 12) + ((y + oy) << 6) + (x + ox);
      labs[cnt++] = lin[sn];
    }
  }
  int bestc = 0, bestl = 0x7fffffff;
  for (int i = 0; i < cnt; ++i) {
    int li = labs[i], c = 0;
    for (int j = 0; j < cnt; ++j) c += (labs[j] == li);
    if (c > bestc || (c == bestc && li < bestl)) { bestc = c; bestl = li; }
  }
  lout[nidx] = (bestc > 0) ? bestl : lin[nidx];
}

__global__ void segmax_kernel(const float* __restrict__ nodes, const int* __restrict__ lab,
                              unsigned* __restrict__ cenc, int* __restrict__ cnt) {
  int nidx = blockIdx.x * blockDim.x + threadIdx.x;
  if (nidx >= NPIX) return;
  int l = lab[nidx];
  atomicAdd(&cnt[l], 1);
#pragma unroll
  for (int c = 0; c < 32; ++c)
    atomicMax(&cenc[(size_t)l * 32 + c], encf(nodes[(size_t)nidx * 32 + c]));
}

__global__ void finalize_kernel(const unsigned* __restrict__ cenc, const int* __restrict__ cnt,
                                float* __restrict__ outc) {
  int idx = blockIdx.x * blockDim.x + threadIdx.x;
  if (idx >= NPIX * 32) return;
  int seg = idx >> 5;
  outc[idx] = (cnt[seg] > 0) ? decf(cenc[idx]) : 0.f;
}

__global__ void labmask_kernel(const int* __restrict__ lab, const int* __restrict__ cnt,
                               int* __restrict__ outi) {
  int nidx = blockIdx.x * blockDim.x + threadIdx.x;
  if (nidx >= NPIX) return;
  outi[NPIX * 32 + nidx] = lab[nidx];
  outi[NPIX * 33 + nidx] = (cnt[nidx] > 0) ? 1 : 0;
}

// ================= host side =================
extern "C" void kernel_launch(void* const* d_in, const int* in_sizes, int n_in,
                              void* d_out, int out_size, void* d_ws, size_t ws_size,
                              hipStream_t stream) {
  const float* img  = (const float*)d_in[0];
  const int*   esrc = (const int*)d_in[1];
  const int*   edst = (const int*)d_in[2];
  const int E = in_sizes[1];

  // params flattened in dict insertion order, starting at d_in[3]
  auto P = [&](int i) { return (const float*)d_in[3 + i]; };
  // 0 sfe1_w, 1 sfe1_b, 2 sfe2_w, 3 sfe2_b
  // 4 + r*8 : rdb{r}: c0_w,c0_b,c1_w,c1_b,c2_w,c2_b,lff_w,lff_b
  // 36 gff1_w, 37 gff1_b, 38 gff2_w, 39 gff2_b

  char* base = (char*)d_ws;
  size_t cur = 0;
  auto alloc = [&](size_t bytes) -> void* {
    cur = (cur + 255) & ~(size_t)255;
    void* r = base + cur;
    cur += bytes;
    return r;
  };

  u16* XB    = (u16*)alloc((size_t)3   * NPIX * 2);
  u16* F1    = (u16*)alloc((size_t)32  * NPIX * 2);
  u16* H80   = (u16*)alloc((size_t)80  * NPIX * 2);
  u16* HIN   = (u16*)alloc((size_t)32  * NPIX * 2);
  u16* GFFIN = (u16*)alloc((size_t)128 * NPIX * 2);
  u16* TMP   = (u16*)alloc((size_t)32  * NPIX * 2);
  float*    NODES  = (float*)alloc((size_t)NPIX * 32 * 4);
  float*    AFF    = (float*)alloc((size_t)E * 4);
  float*    SUMAFF = (float*)alloc((size_t)NPIX * 4);
  int*      DEGI   = (int*)alloc((size_t)NPIX * 4);
  float*    MEANV  = (float*)alloc((size_t)NPIX * 4);
  unsigned* NMASK  = (unsigned*)alloc((size_t)NPIX * 4);
  int*      LABA   = (int*)alloc((size_t)NPIX * 4);
  int*      LABB   = (int*)alloc((size_t)NPIX * 4);
  unsigned* CENC   = (unsigned*)alloc((size_t)NPIX * 32 * 4);
  int*      CNT    = (int*)alloc((size_t)NPIX * 4);

  auto pack = [&](const float* w, int Cout, int Cin, int ks) -> u16* {
    int K = Cin * ks * ks, KT = (K + 31) / 32, MT = (Cout + 15) / 16;
    int total = MT * KT * 512;
    u16* dst = (u16*)alloc((size_t)total * 2);
    pack_w_kernel<<<(total + 255) / 256, 256, 0, stream>>>(w, dst, Cout, Cin, ks, total);
    return dst;
  };
  auto conv = [&](const u16* in, const u16* wp, const float* bias, u16* out,
                  const u16* res, float* nodes, int Cin, int Cout, int ks, int relu) {
    dim3 g(1024, (Cout + 15) / 16);
    if (ks == 3) {
      if (Cin == 3)
        conv_wmma_t<3, 3><<<g, 128, 0, stream>>>(in, wp, bias, out, res, nodes, Cout, relu);
      else if (Cin == 32)
        conv_wmma_t<3, 32><<<g, 128, 0, stream>>>(in, wp, bias, out, res, nodes, Cout, relu);
      else if (Cin == 48)
        conv_wmma_t<3, 48><<<g, 128, 0, stream>>>(in, wp, bias, out, res, nodes, Cout, relu);
      else if (Cin == 64)
        conv_wmma_t<3, 64><<<g, 128, 0, stream>>>(in, wp, bias, out, res, nodes, Cout, relu);
    } else {
      if (Cin == 80)
        conv_wmma_t<1, 80><<<g, 128, 0, stream>>>(in, wp, bias, out, res, nodes, Cout, relu);
      else if (Cin == 128)
        conv_wmma_t<1, 128><<<g, 128, 0, stream>>>(in, wp, bias, out, res, nodes, Cout, relu);
    }
  };

  // pack all weights (deterministic, every call)
  u16* w_sfe1 = pack(P(0), 32, 3, 3);
  u16* w_sfe2 = pack(P(2), 32, 32, 3);
  u16* w_c0[4]; u16* w_c1[4]; u16* w_c2[4]; u16* w_lff[4];
  for (int r = 0; r < 4; ++r) {
    int pb = 4 + r * 8;
    w_c0[r]  = pack(P(pb + 0), 16, 32, 3);
    w_c1[r]  = pack(P(pb + 2), 16, 48, 3);
    w_c2[r]  = pack(P(pb + 4), 16, 64, 3);
    w_lff[r] = pack(P(pb + 6), 32, 80, 1);
  }
  u16* w_gff1 = pack(P(36), 32, 128, 1);
  u16* w_gff2 = pack(P(38), 32, 32, 3);

  // RDN forward
  img2act_kernel<<<(3 * NPIX + 255) / 256, 256, 0, stream>>>(img, XB);
  conv(XB, w_sfe1, P(1), F1,  nullptr, nullptr, 3,  32, 3, 0);
  conv(F1, w_sfe2, P(3), H80, nullptr, nullptr, 32, 32, 3, 0);
  for (int r = 0; r < 4; ++r) {
    int pb = 4 + r * 8;
    hipMemcpyAsync(HIN, H80, (size_t)32 * NPIX * 2, hipMemcpyDeviceToDevice, stream);
    conv(H80, w_c0[r], P(pb + 1), H80 + (size_t)32 * NPIX, nullptr, nullptr, 32, 16, 3, 1);
    conv(H80, w_c1[r], P(pb + 3), H80 + (size_t)48 * NPIX, nullptr, nullptr, 48, 16, 3, 1);
    conv(H80, w_c2[r], P(pb + 5), H80 + (size_t)64 * NPIX, nullptr, nullptr, 64, 16, 3, 1);
    conv(H80, w_lff[r], P(pb + 7), GFFIN + (size_t)r * 32 * NPIX, HIN, nullptr, 80, 32, 1, 0);
    hipMemcpyAsync(H80, GFFIN + (size_t)r * 32 * NPIX, (size_t)32 * NPIX * 2,
                   hipMemcpyDeviceToDevice, stream);
  }
  conv(GFFIN, w_gff1, P(37), TMP, nullptr, nullptr, 128, 32, 1, 0);
  conv(TMP,   w_gff2, P(39), H80, F1, NODES, 32, 32, 3, 0);   // feats + fp32 nodes

  // graph stage
  hipMemsetAsync(SUMAFF, 0, (size_t)NPIX * 4, stream);
  hipMemsetAsync(DEGI,   0, (size_t)NPIX * 4, stream);
  hipMemsetAsync(NMASK,  0, (size_t)NPIX * 4, stream);
  hipMemsetAsync(CENC,   0, (size_t)NPIX * 32 * 4, stream);
  hipMemsetAsync(CNT,    0, (size_t)NPIX * 4, stream);

  int eblocks = (E + 255) / 256;
  aff_kernel<<<eblocks, 256, 0, stream>>>(NODES, esrc, edst, AFF, SUMAFF, DEGI, E);
  mean_kernel<<<NPIX / 256, 256, 0, stream>>>(SUMAFF, DEGI, MEANV);
  keep_kernel<<<eblocks, 256, 0, stream>>>(AFF, MEANV, esrc, edst, NMASK, E);

  iota_kernel<<<NPIX / 256, 256, 0, stream>>>(LABA);
  int* curL = LABA;
  int* nxtL = LABB;
  for (int it = 0; it < 70; ++it) {
    lp_kernel<<<NPIX / 256, 256, 0, stream>>>(curL, nxtL, NMASK);
    int* t = curL; curL = nxtL; nxtL = t;
  }

  segmax_kernel<<<NPIX / 256, 256, 0, stream>>>(NODES, curL, CENC, CNT);
  finalize_kernel<<<(NPIX * 32 + 255) / 256, 256, 0, stream>>>(CENC, CNT, (float*)d_out);
  labmask_kernel<<<NPIX / 256, 256, 0, stream>>>(curL, CNT, (int*)d_out);
}
